// MGNNs_79164837200750
// MI455X (gfx1250) — compile-verified, matrix operand order checked by reading, not run on hardware
//
#include <hip/hip_runtime.h>
#include <math.h>

// ---------------- types ----------------
typedef __bf16 bf16_t;
typedef __attribute__((ext_vector_type(16))) __bf16 v16bf;
typedef __attribute__((ext_vector_type(8)))  __bf16 v8bf;
typedef __attribute__((ext_vector_type(8)))  float  v8f;

#if __has_builtin(__builtin_amdgcn_sched_barrier)
#define SCHED_FENCE() __builtin_amdgcn_sched_barrier(0)
#else
#define SCHED_FENCE() ((void)0)
#endif

// ---------------- sizes ----------------
#define BB     32
#define NOPS   1000
#define OPAD   1024      // K padded for WMMA (zeros)
#define ROWPAD 1008      // M padded for WMMA A rows (zeros)
#define NMAS   20
#define NJOBS  50
#define HID    128
#define OUTOP  8
#define OUTMA  8
#define CATD   512

__device__ __forceinline__ float eluf(float x) { return x > 0.f ? x : (expf(x) - 1.f); }

__device__ __forceinline__ v16bf ld16(const bf16_t* p0, const bf16_t* p1) {
    v8bf lo = *(const v8bf*)p0;
    v8bf hi = *(const v8bf*)p1;
    v16bf r;
#pragma unroll
    for (int e = 0; e < 8; ++e) { r[e] = lo[e]; r[8 + e] = hi[e]; }
    return r;
}

// ---------------- conversion kernels ----------------
// f32 [B,1000,1000] adjacency -> bf16 [B,1008,1024] zero-padded
__global__ void k_cvt_adj(const float* __restrict__ src, bf16_t* __restrict__ dst) {
    long i = (long)blockIdx.x * blockDim.x + threadIdx.x;
    long total = (long)BB * ROWPAD * OPAD;
    if (i >= total) return;
    int k = (int)(i & (OPAD - 1));
    long t = i >> 10;
    int o = (int)(t % ROWPAD);
    int b = (int)(t / ROWPAD);
    float v = 0.f;
    if (o < NOPS && k < NOPS) v = src[((long)b * NOPS + o) * NOPS + k];
    dst[i] = (bf16_t)v;
}

// c1 W [512,128] f32 -> W^T bf16 [128][512]
__global__ void k_cvt_w1t(const float* __restrict__ W, bf16_t* __restrict__ out) {
    int i = blockIdx.x * 256 + threadIdx.x;  // 65536
    int h = i >> 9, k = i & 511;
    out[i] = (bf16_t)W[k * HID + h];
}

// ---------------- op messages: pre / sub / slf ----------------
// layout: i = ((b*128 + h)*1024 + o) so msgT writes are the natural index
__global__ void k_msgs(const float* __restrict__ hin, int din,
                       const float* __restrict__ preW, const float* __restrict__ preb,
                       const float* __restrict__ subW, const float* __restrict__ subb,
                       const float* __restrict__ slfW, const float* __restrict__ slfb,
                       bf16_t* __restrict__ msgPT, bf16_t* __restrict__ msgST,
                       bf16_t* __restrict__ catb) {
    long i = (long)blockIdx.x * 256 + threadIdx.x;  // B*128*1024
    int o = (int)(i & 1023);
    int h = (int)((i >> 10) & 127);
    int b = (int)(i >> 17);
    if (o >= NOPS) { msgPT[i] = (bf16_t)0.f; msgST[i] = (bf16_t)0.f; return; }
    const float* x = hin + ((long)b * NOPS + o) * din;
    float dp = preb[h], ds = subb[h], dl = slfb[h];
    for (int k = 0; k < din; ++k) {
        float xv = x[k];
        dp += xv * preW[k * HID + h];
        ds += xv * subW[k * HID + h];
        dl += xv * slfW[k * HID + h];
    }
    msgPT[i] = (bf16_t)eluf(dp);
    msgST[i] = (bf16_t)eluf(ds);
    catb[((long)b * NOPS + o) * CATD + 384 + h] = (bf16_t)eluf(dl);
}

// ---------------- edge-conditioned machine->op messages + aggregation ----------------
__global__ void k_aggma(const float* __restrict__ hmas, int dinm,
                        const float* __restrict__ edge, const float* __restrict__ adjma,
                        const float* __restrict__ W, const float* __restrict__ bias,
                        bf16_t* __restrict__ catb) {
    int bo = blockIdx.x;
    int b = bo / NOPS, o = bo - b * NOPS;
    int h = threadIdx.x;  // 128
    int dp = dinm + 3;
    __shared__ float sW[11 * 128];
    __shared__ float sB[128];
    __shared__ float sx[20 * 11];
    __shared__ float sadj[20];
    for (int i = h; i < dp * 128; i += 128) sW[i] = W[i];
    sB[h] = bias[h];
    for (int i = h; i < 20 * dp; i += 128) {
        int m = i / dp, k = i - m * dp;
        sx[i] = (k < dinm) ? hmas[((long)b * NMAS + m) * dinm + k]
                           : edge[(((long)b * NOPS + o) * NMAS + m) * 3 + (k - dinm)];
    }
    if (h < NMAS) sadj[h] = adjma[((long)b * NOPS + o) * NMAS + h];
    __syncthreads();
    float acc = 0.f;
    for (int m = 0; m < NMAS; ++m) {
        float d = sB[h];
        const float* xr = sx + m * dp;
        for (int k = 0; k < dp; ++k) d += xr[k] * sW[k * 128 + h];
        acc += sadj[m] * eluf(d);
    }
    catb[((long)b * NOPS + o) * CATD + h] = (bf16_t)acc;
}

// ---------------- WMMA GEMM: adjacency @ msg  (agg_pre / agg_sub) ----------------
// two waves per (batch, 16-row M-tile); each wave owns 4 N-tiles (16x64 strip).
// single-buffer, intra-iteration batching: all 10 fragment loads are issued as
// one clause, a sched fence stops the scheduler sinking them, then the 4-WMMA
// chain runs with graduated s_wait_loadcnt. No loop-carried fragment values ->
// no phi copies, no spills, ~130 VGPRs -> high occupancy.
__global__ void k_gemm_adj(const bf16_t* __restrict__ adj, const bf16_t* __restrict__ msgT,
                           bf16_t* __restrict__ catb, int slotOff) {
    int wid = (blockIdx.x << 3) + (threadIdx.x >> 5);  // 32*63*2 = 4032 waves
    int lane = threadIdx.x & 31;
    int nh = wid & 1;          // which 64-col half
    int rw = wid >> 1;         // 0..2015
    int b = rw / 63, mt = rw - b * 63;
    int lr = lane & 15, hi = lane >> 4;
    const bf16_t* Ar = adj + ((long)b * ROWPAD + mt * 16 + lr) * OPAD + hi * 8;
    const bf16_t* Bb = msgT + (long)b * HID * OPAD + (long)(nh * 64 + lr) * OPAD + hi * 8;
    v8f acc[4];
#pragma unroll
    for (int t = 0; t < 4; ++t) {
#pragma unroll
        for (int r = 0; r < 8; ++r) acc[t][r] = 0.f;
    }
#pragma unroll 1
    for (int k0 = 0; k0 < OPAD; k0 += 32) {
        v16bf a = ld16(Ar + k0, Ar + k0 + 16);
        v16bf b0, b1, b2, b3;
        {
            const bf16_t* bp = Bb + k0;
            b0 = ld16(bp, bp + 16);
            bp += (long)16 * OPAD; b1 = ld16(bp, bp + 16);
            bp += (long)16 * OPAD; b2 = ld16(bp, bp + 16);
            bp += (long)16 * OPAD; b3 = ld16(bp, bp + 16);
        }
        __builtin_prefetch(Ar + k0 + 128, 0, 1);  // stream-ahead on the unique A rows
        SCHED_FENCE();                            // keep the load clause above the WMMAs
        acc[0] = __builtin_amdgcn_wmma_f32_16x16x32_bf16(false, a, false, b0, (short)0, acc[0], false, false);
        acc[1] = __builtin_amdgcn_wmma_f32_16x16x32_bf16(false, a, false, b1, (short)0, acc[1], false, false);
        acc[2] = __builtin_amdgcn_wmma_f32_16x16x32_bf16(false, a, false, b2, (short)0, acc[2], false, false);
        acc[3] = __builtin_amdgcn_wmma_f32_16x16x32_bf16(false, a, false, b3, (short)0, acc[3], false, false);
    }
#pragma unroll
    for (int t = 0; t < 4; ++t) {
        int col = slotOff + nh * 64 + t * 16 + lr;
#pragma unroll
        for (int r = 0; r < 8; ++r) {
            int row = mt * 16 + r + hi * 8;
            if (row < NOPS) catb[((long)b * NOPS + row) * CATD + col] = (bf16_t)acc[t][r];
        }
    }
}

// ---------------- WMMA GEMM: cat[32000x512] @ c1[512x128] -> elu -> h_elu ----------------
__global__ void k_gemm_c1(const bf16_t* __restrict__ catb, const bf16_t* __restrict__ w1t,
                          const float* __restrict__ b1v, float* __restrict__ helu) {
    int wid = (blockIdx.x << 3) + (threadIdx.x >> 5);  // 2000*2 = 4000 waves
    int lane = threadIdx.x & 31;
    int nh = wid & 1;
    int strip = wid >> 1;  // 0..1999
    int lr = lane & 15, hi = lane >> 4;
    const bf16_t* Ar = catb + ((long)strip * 16 + lr) * CATD + hi * 8;
    const bf16_t* Bb = w1t + (long)(nh * 64 + lr) * CATD + hi * 8;
    v8f acc[4];
#pragma unroll
    for (int t = 0; t < 4; ++t) {
#pragma unroll
        for (int r = 0; r < 8; ++r) acc[t][r] = 0.f;
    }
#pragma unroll 1
    for (int k0 = 0; k0 < CATD; k0 += 32) {
        v16bf a = ld16(Ar + k0, Ar + k0 + 16);
        v16bf b0, b1, b2, b3;
        {
            const bf16_t* bp = Bb + k0;
            b0 = ld16(bp, bp + 16);
            bp += (long)16 * CATD; b1 = ld16(bp, bp + 16);
            bp += (long)16 * CATD; b2 = ld16(bp, bp + 16);
            bp += (long)16 * CATD; b3 = ld16(bp, bp + 16);
        }
        __builtin_prefetch(Ar + k0 + 128, 0, 1);
        SCHED_FENCE();
        acc[0] = __builtin_amdgcn_wmma_f32_16x16x32_bf16(false, a, false, b0, (short)0, acc[0], false, false);
        acc[1] = __builtin_amdgcn_wmma_f32_16x16x32_bf16(false, a, false, b1, (short)0, acc[1], false, false);
        acc[2] = __builtin_amdgcn_wmma_f32_16x16x32_bf16(false, a, false, b2, (short)0, acc[2], false, false);
        acc[3] = __builtin_amdgcn_wmma_f32_16x16x32_bf16(false, a, false, b3, (short)0, acc[3], false, false);
    }
#pragma unroll
    for (int t = 0; t < 4; ++t) {
        int col = nh * 64 + t * 16 + lr;
        float bb = b1v[col];
#pragma unroll
        for (int r = 0; r < 8; ++r) {
            long row = (long)strip * 16 + r + hi * 8;
            helu[row * HID + col] = eluf(acc[t][r] + bb);
        }
    }
}

// ---------------- c2: h_elu[32000x128] @ W2[128x8] -> h_ops ----------------
__global__ void k_c2(const float* __restrict__ helu, const float* __restrict__ W2,
                     const float* __restrict__ b2, float* __restrict__ hops) {
    int i = blockIdx.x * 256 + threadIdx.x;  // 32000*8
    int row = i >> 3, j = i & 7;
    float a = b2[j];
    const float* hr = helu + (long)row * HID;
    for (int h = 0; h < HID; ++h) a += hr[h] * W2[h * OUTOP + j];
    hops[i] = a;
}

// ---------------- machine update: msg_op ----------------
__global__ void k_msgop(const float* __restrict__ hops, const float* __restrict__ aggW,
                        const float* __restrict__ aggb, float* __restrict__ msgop) {
    int i = blockIdx.x * 256 + threadIdx.x;  // 32000*128
    int row = i >> 7, h = i & 127;
    float d = aggb[h];
    const float* hr = hops + (long)row * OUTOP;
    for (int k = 0; k < OUTOP; ++k) d += hr[k] * aggW[k * HID + h];
    msgop[i] = eluf(d);
}

// ---------------- machine update: aggregate ops per machine ----------------
__global__ void k_aggop(const float* __restrict__ adjma, const float* __restrict__ msgop,
                        float* __restrict__ catm) {
    int b = blockIdx.x / NMAS, m = blockIdx.x % NMAS;
    int h = threadIdx.x;  // 128
    __shared__ float sadj[128];
    float acc = 0.f;
    for (int o0 = 0; o0 < NOPS; o0 += 128) {
        int o = o0 + h;
        sadj[h] = (o < NOPS) ? adjma[((long)b * NOPS + o) * NMAS + m] : 0.f;
        __syncthreads();
        int lim = NOPS - o0; if (lim > 128) lim = 128;
        for (int oo = 0; oo < lim; ++oo)
            acc += sadj[oo] * msgop[((long)b * NOPS + o0 + oo) * HID + h];
        __syncthreads();
    }
    catm[((long)b * NMAS + m) * 256 + h] = acc;
}

// ---------------- machine update: slf + c1 + c2 ----------------
__global__ void k_maupd(const float* __restrict__ hmin, int dins,
                        const float* __restrict__ slfW, const float* __restrict__ slfb,
                        const float* __restrict__ c1W, const float* __restrict__ c1b,
                        const float* __restrict__ c2W, const float* __restrict__ c2b,
                        const float* __restrict__ catm, float* __restrict__ hmout) {
    int b = blockIdx.x / NMAS, m = blockIdx.x % NMAS;
    int h = threadIdx.x;  // 128
    __shared__ float sin[256];
    __shared__ float sh[128];
    float d = slfb[h];
    const float* x = hmin + ((long)b * NMAS + m) * dins;
    for (int k = 0; k < dins; ++k) d += x[k] * slfW[k * HID + h];
    sin[128 + h] = eluf(d);
    sin[h] = catm[((long)b * NMAS + m) * 256 + h];
    __syncthreads();
    float e = c1b[h];
    for (int k = 0; k < 256; ++k) e += sin[k] * c1W[k * HID + h];
    sh[h] = eluf(e);
    __syncthreads();
    if (h < OUTMA) {
        float f = c2b[h];
        for (int k = 0; k < HID; ++k) f += sh[k] * c2W[k * OUTMA + h];
        hmout[((long)b * NMAS + m) * OUTMA + h] = f;
    }
}

// ---------------- pooling ----------------
__global__ void k_pool(const float* __restrict__ hops, const float* __restrict__ hmas,
                       float* __restrict__ pop, float* __restrict__ pma) {
    int b = blockIdx.x, t = threadIdx.x;  // 256
    __shared__ float s[8 * 256];
    float a[8];
#pragma unroll
    for (int j = 0; j < 8; ++j) a[j] = 0.f;
    for (int o = t; o < NOPS; o += 256) {
        const float* r = hops + ((long)b * NOPS + o) * OUTOP;
#pragma unroll
        for (int j = 0; j < 8; ++j) a[j] += r[j];
    }
#pragma unroll
    for (int j = 0; j < 8; ++j) s[j * 256 + t] = a[j];
    __syncthreads();
    for (int st = 128; st > 0; st >>= 1) {
        if (t < st) {
#pragma unroll
            for (int j = 0; j < 8; ++j) s[j * 256 + t] += s[j * 256 + t + st];
        }
        __syncthreads();
    }
    if (t < 8) pop[b * 8 + t] = s[t * 256] * (1.f / NOPS);
    if (t >= 8 && t < 16) {
        int j = t - 8;
        float mm = 0.f;
        for (int m = 0; m < NMAS; ++m) mm += hmas[((long)b * NMAS + m) * OUTMA + j];
        pma[b * 8 + j] = mm * (1.f / NMAS);
    }
}

// ---------------- actor MLP + mask ----------------
__global__ void k_actor(const float* __restrict__ hops, const float* __restrict__ hmas,
                        const float* __restrict__ pop, const float* __restrict__ pma,
                        const int* __restrict__ jobs, const unsigned char* __restrict__ elig,
                        const float* __restrict__ W0, const float* __restrict__ B0,
                        const float* __restrict__ W1, const float* __restrict__ B1,
                        const float* __restrict__ W2, const float* __restrict__ B2,
                        const float* __restrict__ W3, const float* __restrict__ B3,
                        float* __restrict__ scores) {
    __shared__ float sW0[32 * 64], sW1[64 * 64], sW2[64 * 64], sW3[64];
    __shared__ float sB0[64], sB1[64], sB2[64];
    int tid = threadIdx.x;
    for (int i = tid; i < 32 * 64; i += 256) sW0[i] = W0[i];
    for (int i = tid; i < 64 * 64; i += 256) { sW1[i] = W1[i]; sW2[i] = W2[i]; }
    for (int i = tid; i < 64; i += 256) { sW3[i] = W3[i]; sB0[i] = B0[i]; sB1[i] = B1[i]; sB2[i] = B2[i]; }
    __syncthreads();
    int idx = blockIdx.x * 256 + tid;  // 32000 exactly
    int b = idx / 1000, r = idx - b * 1000;
    int m = r / NJOBS, j = r - m * NJOBS;
    float x[32];
    int op = jobs[b * NJOBS + j];
    const float* hj = hops + ((long)b * NOPS + op) * OUTOP;
    const float* hm = hmas + ((long)b * NMAS + m) * OUTMA;
#pragma unroll
    for (int k = 0; k < 8; ++k) {
        x[k] = hj[k]; x[8 + k] = hm[k]; x[16 + k] = pop[b * 8 + k]; x[24 + k] = pma[b * 8 + k];
    }
    float y[64], z[64];
    for (int h = 0; h < 64; ++h) {
        float d = sB0[h];
        for (int k = 0; k < 32; ++k) d += x[k] * sW0[k * 64 + h];
        y[h] = tanhf(d);
    }
    for (int h = 0; h < 64; ++h) {
        float d = sB1[h];
        for (int k = 0; k < 64; ++k) d += y[k] * sW1[k * 64 + h];
        z[h] = tanhf(d);
    }
    for (int h = 0; h < 64; ++h) {
        float d = sB2[h];
        for (int k = 0; k < 64; ++k) d += z[k] * sW2[k * 64 + h];
        y[h] = tanhf(d);
    }
    float sc = B3[0];
    for (int k = 0; k < 64; ++k) sc += y[k] * sW3[k];
    bool e = elig[((long)b * NJOBS + j) * NMAS + m] != 0;
    scores[idx] = e ? sc : -1e9f;
}

// ---------------- softmax over M*J per batch ----------------
__global__ void k_softmax(const float* __restrict__ scores, float* __restrict__ out) {
    int b = blockIdx.x, t = threadIdx.x;  // 256
    __shared__ float red[256];
    const float* s = scores + (long)b * 1000;
    float mx = -3.0e38f;
    for (int i = t; i < 1000; i += 256) mx = fmaxf(mx, s[i]);
    red[t] = mx; __syncthreads();
    for (int st = 128; st > 0; st >>= 1) { if (t < st) red[t] = fmaxf(red[t], red[t + st]); __syncthreads(); }
    float M = red[0]; __syncthreads();
    float sm = 0.f;
    for (int i = t; i < 1000; i += 256) sm += expf(s[i] - M);
    red[t] = sm; __syncthreads();
    for (int st = 128; st > 0; st >>= 1) { if (t < st) red[t] += red[t + st]; __syncthreads(); }
    float S = red[0];
    float inv = 1.f / S;
    for (int i = t; i < 1000; i += 256) out[(long)b * 1000 + i] = expf(s[i] - M) * inv;
}

// ---------------- host ----------------
extern "C" void kernel_launch(void* const* d_in, const int* in_sizes, int n_in,
                              void* d_out, int out_size, void* d_ws, size_t ws_size,
                              hipStream_t stream) {
    const float* op_ma_adj  = (const float*)d_in[0];
    const float* op_pre_adj = (const float*)d_in[1];
    const float* op_sub_adj = (const float*)d_in[2];
    const float* raw_ops    = (const float*)d_in[3];
    const float* raw_mas    = (const float*)d_in[4];
    const float* raw_edge   = (const float*)d_in[5];
    const int*   jobs_idx   = (const int*)d_in[6];
    const unsigned char* eligible = (const unsigned char*)d_in[7];

    // params flattened in jax dict-key alphabetical order: actor, gat, mlp
    const float *aW[4], *aB[4];
    for (int l = 0; l < 4; ++l) { aW[l] = (const float*)d_in[8 + 2 * l]; aB[l] = (const float*)d_in[9 + 2 * l]; }
    struct Gat { const float *c1W, *c1b, *c2W, *c2b, *maW, *mab, *preW, *preb, *slfW, *slfb, *subW, *subb; } g[3];
    for (int l = 0; l < 3; ++l) {
        int p = 16 + 12 * l;
        g[l].c1W = (const float*)d_in[p + 0];  g[l].c1b = (const float*)d_in[p + 1];
        g[l].c2W = (const float*)d_in[p + 2];  g[l].c2b = (const float*)d_in[p + 3];
        g[l].maW = (const float*)d_in[p + 4];  g[l].mab = (const float*)d_in[p + 5];
        g[l].preW = (const float*)d_in[p + 6]; g[l].preb = (const float*)d_in[p + 7];
        g[l].slfW = (const float*)d_in[p + 8]; g[l].slfb = (const float*)d_in[p + 9];
        g[l].subW = (const float*)d_in[p + 10]; g[l].subb = (const float*)d_in[p + 11];
    }
    struct Mlp { const float *aggW, *aggb, *c1W, *c1b, *c2W, *c2b, *slfW, *slfb; } q[3];
    for (int l = 0; l < 3; ++l) {
        int p = 52 + 8 * l;
        q[l].aggW = (const float*)d_in[p + 0]; q[l].aggb = (const float*)d_in[p + 1];
        q[l].c1W = (const float*)d_in[p + 2];  q[l].c1b = (const float*)d_in[p + 3];
        q[l].c2W = (const float*)d_in[p + 4];  q[l].c2b = (const float*)d_in[p + 5];
        q[l].slfW = (const float*)d_in[p + 6]; q[l].slfb = (const float*)d_in[p + 7];
    }

    // workspace carve
    char* w = (char*)d_ws;
    size_t used = 0;
    auto carve = [&](size_t bytes) -> void* {
        void* p = w + used;
        used += (bytes + 255) & ~(size_t)255;
        return p;
    };
    const size_t ADJ_E  = (size_t)BB * ROWPAD * OPAD;       // bf16 elems
    const size_t MSGT_E = (size_t)BB * HID * OPAD;
    const size_t CAT_E  = (size_t)BB * NOPS * CATD;
    bf16_t* adjP  = (bf16_t*)carve(ADJ_E * 2);
    bf16_t* adjS  = (bf16_t*)carve(ADJ_E * 2);
    bf16_t* msgPT = (bf16_t*)carve(MSGT_E * 2);
    bf16_t* msgST = (bf16_t*)carve(MSGT_E * 2);
    bf16_t* catb  = (bf16_t*)carve(CAT_E * 2);
    float*  helu  = (float*)carve((size_t)BB * NOPS * HID * 4);   // reused as msg_op
    float*  h_ops = (float*)carve((size_t)BB * NOPS * OUTOP * 4);
    float*  h_mas = (float*)carve((size_t)BB * NMAS * OUTMA * 4);
    float*  catm  = (float*)carve((size_t)BB * NMAS * 256 * 4);
    float*  pop   = (float*)carve((size_t)BB * 8 * 4);
    float*  pma   = (float*)carve((size_t)BB * 8 * 4);
    float*  scor  = (float*)carve((size_t)BB * 1000 * 4);
    bf16_t* w1t   = (bf16_t*)carve((size_t)HID * CATD * 2);
    if (used > ws_size) return;  // workspace too small: bail deterministically

    // one-time adjacency bf16 conversion (fits in L2 for the 3-layer GEMM reuse)
    {
        int blocks = (int)((ADJ_E + 255) / 256);
        k_cvt_adj<<<blocks, 256, 0, stream>>>(op_pre_adj, adjP);
        k_cvt_adj<<<blocks, 256, 0, stream>>>(op_sub_adj, adjS);
    }

    for (int l = 0; l < 3; ++l) {
        const float* hin_ops = (l == 0) ? raw_ops : h_ops;
        int din_op = (l == 0) ? 6 : 8;
        const float* hin_mas = (l == 0) ? raw_mas : h_mas;
        int din_ma = (l == 0) ? 3 : 8;

        k_msgs<<<(BB * HID * OPAD) / 256, 256, 0, stream>>>(
            hin_ops, din_op, g[l].preW, g[l].preb, g[l].subW, g[l].subb,
            g[l].slfW, g[l].slfb, msgPT, msgST, catb);
        k_aggma<<<BB * NOPS, 128, 0, stream>>>(
            hin_mas, din_ma, raw_edge, op_ma_adj, g[l].maW, g[l].mab, catb);
        k_gemm_adj<<<(BB * 63 * 2) / 8, 256, 0, stream>>>(adjP, msgPT, catb, 128);
        k_gemm_adj<<<(BB * 63 * 2) / 8, 256, 0, stream>>>(adjS, msgST, catb, 256);
        k_cvt_w1t<<<(HID * CATD) / 256, 256, 0, stream>>>(g[l].c1W, w1t);
        k_gemm_c1<<<(BB * NOPS / 16 * 2) / 8, 256, 0, stream>>>(catb, w1t, g[l].c1b, helu);
        k_c2<<<(BB * NOPS * OUTOP) / 256, 256, 0, stream>>>(helu, g[l].c2W, g[l].c2b, h_ops);

        k_msgop<<<(BB * NOPS * HID) / 256, 256, 0, stream>>>(h_ops, q[l].aggW, q[l].aggb, helu);
        k_aggop<<<BB * NMAS, 128, 0, stream>>>(op_ma_adj, helu, catm);
        k_maupd<<<BB * NMAS, 128, 0, stream>>>(hin_mas, din_ma, q[l].slfW, q[l].slfb,
                                               q[l].c1W, q[l].c1b, q[l].c2W, q[l].c2b,
                                               catm, h_mas);
    }

    k_pool<<<BB, 256, 0, stream>>>(h_ops, h_mas, pop, pma);
    k_actor<<<(BB * NMAS * NJOBS) / 256, 256, 0, stream>>>(
        h_ops, h_mas, pop, pma, jobs_idx, eligible,
        aW[0], aB[0], aW[1], aB[1], aW[2], aB[2], aW[3], aB[3], scor);
    k_softmax<<<BB, 256, 0, stream>>>(scor, (float*)d_out);
}
